// SSM_Individual_Head_31490700214763
// MI455X (gfx1250) — matrix operand order repaired and checked by reading, not compiled, not run on hardware
//
#include <hip/hip_runtime.h>
#include <hip/hip_bf16.h>
#include <cmath>

typedef __attribute__((ext_vector_type(16))) _Float16 v16h;
typedef __attribute__((ext_vector_type(8)))  float    v8f;

#define BATCH   32
#define LSEQ    4096
#define DIN     3
#define HCH     128            // d_model
#define NSTATE  64
#define NLAY    4
#define DOUT    10
#define MROWS   (BATCH*LSEQ)   // 131072
#define NCH     8              // scan chunks
#define CHT     (LSEQ/NCH)     // 512
#define NTILES  (MROWS/16)     // 8192
#define TPB     (LSEQ/16)      // 256 tiles per batch

// ---- WMMA fragment index helpers (CDNA5 ISA 7.12.2, wave32) ----
// 16-bit A 16x32: lane L: M=L&15; khalf=(L>>4); vgpr v holds 2 halfs:
//   K = (v>>2)*16 + khalf*8 + (v&3)*2 + p
__device__ __forceinline__ int a_k(int lane, int e) {
  int v = e >> 1, p = e & 1;
  return ((v >> 2) << 4) + (((lane >> 4) & 1) << 3) + ((v & 3) << 1) + p;
}
// 16-bit B 32x16: lane L: N=L&15; K = (e>>3)*16 + ((L>>4)&1)*8 + (e&7)
__device__ __forceinline__ int b_k(int lane, int e) {
  return ((e >> 3) << 4) + (((lane >> 4) & 1) << 3) + (e & 7);
}

union AF { v16h v; _Float16 h[16]; };
union CF { v8f  v; float     f[8]; };

// ---- weight packing into B-fragment lane order (f32 -> f16) ----
// 256x128 weight (Wglu / Whead): per layer 4 kb * 16 nt * 32 lanes * 16 elems
__global__ void pack_w256(const float* __restrict__ W, _Float16* __restrict__ P) {
  int idx = blockIdx.x * blockDim.x + threadIdx.x;    // NLAY*32768
  int e    = idx & 15;
  int lane = (idx >> 4) & 31;
  int nt   = (idx >> 9) & 15;
  int kb   = (idx >> 13) & 3;
  int lay  = idx >> 15;
  int k = kb * 32 + b_k(lane, e);
  int o = nt * 16 + (lane & 15);
  P[idx] = (_Float16)W[((size_t)lay * 256 + o) * HCH + k];
}

// 10x128 Wout, N padded to 16: per layer 4 kb * 32 lanes * 16 elems = 2048
__global__ void pack_wout(const float* __restrict__ W, _Float16* __restrict__ P) {
  int idx = blockIdx.x * blockDim.x + threadIdx.x;    // NLAY*2048
  int e    = idx & 15;
  int lane = (idx >> 4) & 31;
  int kb   = (idx >> 9) & 3;
  int lay  = idx >> 11;
  int k = kb * 32 + b_k(lane, e);
  int o = lane & 15;
  P[idx] = (o < DOUT) ? (_Float16)W[((size_t)lay * DOUT + o) * HCH + k]
                      : (_Float16)0.f;
}

// ---- input projection: u[b,l,h] = x[b, l*3 + c] . Win[h,:] + b_in[h] ----
__global__ void input_proj(const float* __restrict__ x, const float* __restrict__ Win,
                           const float* __restrict__ b_in, float* __restrict__ u) {
  int idx = blockIdx.x * blockDim.x + threadIdx.x;    // MROWS*HCH
  int h   = idx & (HCH - 1);
  int row = idx >> 7;
  int b = row >> 12, l = row & (LSEQ - 1);
  const float* xp = x + (size_t)b * (DIN * LSEQ) + (size_t)l * DIN;
  u[idx] = b_in[h] + xp[0] * Win[h * 3 + 0] + xp[1] * Win[h * 3 + 1]
                   + xp[2] * Win[h * 3 + 2];
}

// ---- S4D scan, pass 1: per-(b,h,chunk) local carry (zero initial state) ----
__global__ void __launch_bounds__(HCH, 1)
scan_pass1(const float* __restrict__ u, float* __restrict__ carry,
           const float* __restrict__ log_dt, const float* __restrict__ log_A, int lay) {
  int h = threadIdx.x;
  int b = blockIdx.x / NCH, c = blockIdx.x % NCH;
  float dt = expf(log_dt[lay * HCH + h]);
  float s[NSTATE], Ab[NSTATE];
#pragma unroll
  for (int n = 0; n < NSTATE; ++n) {
    float A = -expf(log_A[((size_t)lay * HCH + h) * NSTATE + n]);
    Ab[n] = expf(dt * A);
    s[n] = 0.f;
  }
  const float* up = u + ((size_t)b * LSEQ + (size_t)c * CHT) * HCH + h;
  for (int l = 0; l < CHT; ++l) {
    float uu = up[(size_t)l * HCH];
#pragma unroll
    for (int n = 0; n < NSTATE; ++n) s[n] = fmaf(s[n], Ab[n], uu);
  }
  float* cp = carry + (((size_t)(b * NCH + c)) * HCH + h) * NSTATE;
#pragma unroll
  for (int n = 0; n < NSTATE; ++n) cp[n] = s[n];
}

// ---- scan pass 2: combine carries across chunks (serial over NCH=8) ----
__global__ void scan_combine(const float* __restrict__ carry, float* __restrict__ Sinit,
                             const float* __restrict__ log_dt,
                             const float* __restrict__ log_A, int lay) {
  int idx = blockIdx.x * blockDim.x + threadIdx.x;    // BATCH*HCH*NSTATE
  int n = idx & (NSTATE - 1);
  int h = (idx >> 6) & (HCH - 1);
  int b = idx >> 13;
  float dt = expf(log_dt[lay * HCH + h]);
  float A  = -expf(log_A[((size_t)lay * HCH + h) * NSTATE + n]);
  float AbT = expf(dt * A * (float)CHT);              // decay over a full chunk
  float S = 0.f;
  for (int c = 0; c < NCH; ++c) {
    size_t off = (((size_t)(b * NCH + c)) * HCH + h) * NSTATE + n;
    Sinit[off] = S;                                   // state entering chunk c
    S = fmaf(S, AbT, carry[off]);
  }
}

// ---- scan pass 3: rescan with true initial state, emit v = y + u*D ----
__global__ void __launch_bounds__(HCH, 1)
scan_pass2(const float* __restrict__ u, const float* __restrict__ Sinit,
           float* __restrict__ v, const float* __restrict__ log_dt,
           const float* __restrict__ log_A, const float* __restrict__ Ck,
           const float* __restrict__ Dp, int lay) {
  int h = threadIdx.x;
  int b = blockIdx.x / NCH, c = blockIdx.x % NCH;
  float dt = expf(log_dt[lay * HCH + h]);
  float D  = Dp[lay * HCH + h];
  float s[NSTATE], Ab[NSTATE], CdB[NSTATE];
  const float* sp = Sinit + (((size_t)(b * NCH + c)) * HCH + h) * NSTATE;
#pragma unroll
  for (int n = 0; n < NSTATE; ++n) {
    float A   = -expf(log_A[((size_t)lay * HCH + h) * NSTATE + n]);
    float dtA = dt * A;
    Ab[n]  = expf(dtA);
    CdB[n] = Ck[((size_t)lay * HCH + h) * NSTATE + n] * (expm1f(dtA) / A);
    s[n]   = sp[n];
  }
  const size_t base = ((size_t)b * LSEQ + (size_t)c * CHT) * HCH + h;
  for (int l = 0; l < CHT; ++l) {
    float uu = u[base + (size_t)l * HCH];
    float y = 0.f;
#pragma unroll
    for (int n = 0; n < NSTATE; ++n) {
      s[n] = fmaf(s[n], Ab[n], uu);
      y    = fmaf(CdB[n], s[n], y);
    }
    v[base + (size_t)l * HCH] = y + uu * D;
  }
}

// ---- fused: gelu(v) @ Wglu^T + bglu -> GLU -> +res -> LayerNorm -> u ----
__global__ void s4out_glu_ln(const float* __restrict__ vin, float* __restrict__ u,
                             const _Float16* __restrict__ Wp,
                             const float* __restrict__ bglu,
                             const float* __restrict__ gamma,
                             const float* __restrict__ beta) {
  __shared__ _Float16 aLds[16 * HCH];   // gelu'd activations, f16
  __shared__ float    zLds[16 * 256];   // GEMM output tile
  int lane = threadIdx.x;
  size_t m0 = (size_t)blockIdx.x * 16;

  for (int t = lane; t < 16 * HCH; t += 32) {
    float xv = vin[(m0 + (t >> 7)) * HCH + (t & 127)];
    aLds[t] = (_Float16)(0.5f * xv * (1.f + erff(xv * 0.70710678118654752f)));
  }
  __syncthreads();

  AF a[4];
#pragma unroll
  for (int kb = 0; kb < 4; ++kb)
#pragma unroll
    for (int e = 0; e < 16; ++e)
      a[kb].h[e] = aLds[(lane & 15) * HCH + kb * 32 + a_k(lane, e)];

  for (int nt = 0; nt < 16; ++nt) {
    CF c = {};
#pragma unroll
    for (int kb = 0; kb < 4; ++kb) {
      v16h bf = *reinterpret_cast<const v16h*>(Wp + (((kb * 16 + nt) * 32 + lane) << 4));
      c.v = __builtin_amdgcn_wmma_f32_16x16x32_f16(false, a[kb].v, false, bf,
                                                   (short)0, c.v, false, false);
    }
#pragma unroll
    for (int r = 0; r < 8; ++r)
      zLds[(r + 8 * (lane >> 4)) * 256 + nt * 16 + (lane & 15)] = c.f[r];
  }
  __syncthreads();

  if (lane < 16) {
    int row = lane;
    float sum = 0.f, sq = 0.f;
    for (int o = 0; o < HCH; ++o) {
      float za = zLds[row * 256 + o] + bglu[o];
      float zb = zLds[row * 256 + 128 + o] + bglu[128 + o];
      float g  = za / (1.f + expf(-zb));                 // GLU
      float t  = u[(m0 + row) * HCH + o] + g;            // residual
      zLds[row * 256 + o] = t;
      sum += t; sq += t * t;
    }
    float mean = sum * (1.f / HCH);
    float var  = sq * (1.f / HCH) - mean * mean;
    float rstd = rsqrtf(var + 1e-5f);
    for (int o = 0; o < HCH; ++o) {
      float t = zLds[row * 256 + o];
      u[(m0 + row) * HCH + o] = (t - mean) * rstd * gamma[o] + beta[o];
    }
  }
}

// ---- fused head: GLU(u@Whead^T+bhead) @ Wout^T, per-tile partial sums ----
__global__ void head_partial(const float* __restrict__ u,
                             const _Float16* __restrict__ Wh,
                             const float* __restrict__ bhead,
                             const _Float16* __restrict__ Wo,
                             float* __restrict__ part) {
  __shared__ _Float16 aLds[16 * HCH];
  __shared__ float    zLds[16 * 256];
  int lane = threadIdx.x;
  size_t m0 = (size_t)blockIdx.x * 16;

  for (int t = lane; t < 16 * HCH; t += 32)
    aLds[t] = (_Float16)u[(m0 + (t >> 7)) * HCH + (t & 127)];
  __syncthreads();

  AF a[4];
#pragma unroll
  for (int kb = 0; kb < 4; ++kb)
#pragma unroll
    for (int e = 0; e < 16; ++e)
      a[kb].h[e] = aLds[(lane & 15) * HCH + kb * 32 + a_k(lane, e)];

  for (int nt = 0; nt < 16; ++nt) {
    CF c = {};
#pragma unroll
    for (int kb = 0; kb < 4; ++kb) {
      v16h bf = *reinterpret_cast<const v16h*>(Wh + (((kb * 16 + nt) * 32 + lane) << 4));
      c.v = __builtin_amdgcn_wmma_f32_16x16x32_f16(false, a[kb].v, false, bf,
                                                   (short)0, c.v, false, false);
    }
#pragma unroll
    for (int r = 0; r < 8; ++r)
      zLds[(r + 8 * (lane >> 4)) * 256 + nt * 16 + (lane & 15)] = c.f[r];
  }
  __syncthreads();

  {  // GLU -> h (f16) back into aLds; 32 lanes cover 16 rows x two 64-col halves
    int row = lane & 15;
    int c0  = (lane >> 4) * 64;
    for (int j = 0; j < 64; ++j) {
      int o = c0 + j;
      float za = zLds[row * 256 + o] + bhead[o];
      float zb = zLds[row * 256 + 128 + o] + bhead[128 + o];
      aLds[row * HCH + o] = (_Float16)(za / (1.f + expf(-zb)));
    }
  }
  __syncthreads();

  AF a2[4];
#pragma unroll
  for (int kb = 0; kb < 4; ++kb)
#pragma unroll
    for (int e = 0; e < 16; ++e)
      a2[kb].h[e] = aLds[(lane & 15) * HCH + kb * 32 + a_k(lane, e)];

  CF c2 = {};
#pragma unroll
  for (int kb = 0; kb < 4; ++kb) {
    v16h bf = *reinterpret_cast<const v16h*>(Wo + (((kb * 32) + lane) << 4));
    c2.v = __builtin_amdgcn_wmma_f32_16x16x32_f16(false, a2[kb].v, false, bf,
                                                  (short)0, c2.v, false, false);
  }
  float srow = 0.f;
#pragma unroll
  for (int r = 0; r < 8; ++r) srow += c2.f[r];   // sum over 8 M-rows this lane holds
  zLds[lane] = srow;
  __syncthreads();
  if (lane < DOUT)
    part[(size_t)blockIdx.x * DOUT + lane] = zLds[lane] + zLds[lane + 16];
}

// ---- final fixed-order reduction: out[i,b,o] = bout + mean over L ----
__global__ void reduce_out(const float* __restrict__ part, const float* __restrict__ bout,
                           float* __restrict__ out) {
  int idx = blockIdx.x * blockDim.x + threadIdx.x;
  if (idx >= NLAY * BATCH * DOUT) return;
  int o = idx % DOUT;
  int b = (idx / DOUT) % BATCH;
  int i = idx / (DOUT * BATCH);
  const float* p = part + (size_t)i * NTILES * DOUT + (size_t)b * TPB * DOUT + o;
  float s = 0.f;
  for (int t = 0; t < TPB; ++t) s += p[(size_t)t * DOUT];
  out[idx] = bout[i * DOUT + o] + s * (1.f / (float)LSEQ);
}

extern "C" void kernel_launch(void* const* d_in, const int* in_sizes, int n_in,
                              void* d_out, int out_size, void* d_ws, size_t ws_size,
                              hipStream_t stream) {
  (void)in_sizes; (void)n_in; (void)out_size; (void)ws_size;
  const float* x      = (const float*)d_in[0];
  const float* Win    = (const float*)d_in[1];
  const float* b_in   = (const float*)d_in[2];
  const float* log_dt = (const float*)d_in[3];
  const float* log_A  = (const float*)d_in[4];
  const float* Ck     = (const float*)d_in[5];
  const float* Dp     = (const float*)d_in[6];
  const float* Wglu   = (const float*)d_in[7];
  const float* bglu   = (const float*)d_in[8];
  const float* gamma  = (const float*)d_in[9];
  const float* beta   = (const float*)d_in[10];
  const float* Whead  = (const float*)d_in[11];
  const float* bhead  = (const float*)d_in[12];
  const float* Wout   = (const float*)d_in[13];
  const float* bout   = (const float*)d_in[14];

  char* ws = (char*)d_ws;
  float*    u     = (float*)(ws + (size_t)0);            // 64 MB
  float*    v     = (float*)(ws + (size_t)67108864);     // 64 MB
  float*    carry = (float*)(ws + (size_t)134217728);    // 8 MB
  float*    Sini  = (float*)(ws + (size_t)142606336);    // 8 MB
  _Float16* pGlu  = (_Float16*)(ws + (size_t)150994944); // 256 KB
  _Float16* pHead = (_Float16*)(ws + (size_t)151257088); // 256 KB
  _Float16* pWo   = (_Float16*)(ws + (size_t)151519232); // 16 KB
  float*    part  = (float*)(ws + (size_t)151535616);    // 1.25 MB

  pack_w256<<<NLAY * 32768 / 256, 256, 0, stream>>>(Wglu, pGlu);
  pack_w256<<<NLAY * 32768 / 256, 256, 0, stream>>>(Whead, pHead);
  pack_wout<<<NLAY * 2048 / 256, 256, 0, stream>>>(Wout, pWo);
  input_proj<<<MROWS * HCH / 256, 256, 0, stream>>>(x, Win, b_in, u);

  for (int i = 0; i < NLAY; ++i) {
    scan_pass1<<<BATCH * NCH, HCH, 0, stream>>>(u, carry, log_dt, log_A, i);
    scan_combine<<<BATCH * HCH * NSTATE / 256, 256, 0, stream>>>(carry, Sini, log_dt, log_A, i);
    scan_pass2<<<BATCH * NCH, HCH, 0, stream>>>(u, Sini, v, log_dt, log_A, Ck, Dp, i);
    s4out_glu_ln<<<NTILES, 32, 0, stream>>>(v, u, pGlu + (size_t)i * 32768,
                                            bglu + i * 256, gamma + i * HCH, beta + i * HCH);
    head_partial<<<NTILES, 32, 0, stream>>>(u, pHead + (size_t)i * 32768,
                                            bhead + i * 256, pWo + (size_t)i * 2048,
                                            part + (size_t)i * NTILES * DOUT);
  }
  reduce_out<<<(NLAY * BATCH * DOUT + 255) / 256, 256, 0, stream>>>(part, bout, (float*)d_out);
}
